// BinaryTreeLstm_44925357916242
// MI455X (gfx1250) — matrix operand order repaired and checked by reading, not compiled, not run on hardware
//
#include <hip/hip_runtime.h>
#include <hip/hip_bf16.h>
#include <cstdint>
#include <cstddef>

// ---------------------------------------------------------------------------
// Types
// ---------------------------------------------------------------------------
typedef __attribute__((ext_vector_type(16))) __bf16   v16bf;
typedef __attribute__((ext_vector_type(8)))  float    v8f;
typedef __attribute__((ext_vector_type(4)))  uint32_t u32x4;

#define N_LEAVES   131072
#define EMBED      300
#define HID        150
#define HPAD       160      // HID padded to 16
#define NLAB       5
#define KT_LEAF    10       // K = 320 (300 padded), 10 k-tiles of 32
#define KT_COMP    10       // K = 320 (2*160), 10 k-tiles of 32
#define NT_LEAF    20       // N = 320 (Wcx 160 | Wox 160) -> 20 n-tiles
#define NT_COMP    40       // N = 640 (4 gates x 160)     -> 40 n-tiles

union FragB16 {
  v16bf bf;
  struct { u32x4 lo, hi; } u;
};

// A-matrix 16x32 bf16 fragment from LDS tile laid out [16 rows][32 k] (64B rows).
// lanes 0-15: row=lane, K in {0..7, 16..23}; lanes 16-31: row=lane-16, K in {8..15, 24..31}
__device__ inline FragB16 load_fragA(const __bf16* tile, int lane) {
  const char* p = (const char*)tile + ((lane & 15) * 64 + ((lane >> 4) * 16));
  FragB16 f;
  f.u.lo = *(const u32x4*)p;
  f.u.hi = *(const u32x4*)(p + 32);
  return f;
}

// B-matrix 32x16 bf16 fragment from LDS tile laid out [16 cols][32 k] (64B rows).
// lanes 0-15: col=lane, K in {0..7, 16..23}; lanes 16-31: col=lane-16, K in {8..15, 24..31}
__device__ inline FragB16 load_fragB(const __bf16* tile, int lane) {
  const char* p = (const char*)tile + ((lane & 15) * 64 + ((lane >> 4) * 16));
  FragB16 f;
  f.u.lo = *(const u32x4*)p;
  f.u.hi = *(const u32x4*)(p + 32);
  return f;
}

__device__ inline float sigf(float x) { return 1.0f / (1.0f + __expf(-x)); }

// Async memory -> LDS (no VGPR round trip); tracked by ASYNCcnt.
// LDS address = low 32 bits of the generic shared-space pointer.
__device__ inline void async_copy_b128(void* ldsDst, const void* gSrc) {
  unsigned l = (unsigned)(uintptr_t)ldsDst;
  asm volatile("global_load_async_to_lds_b128 %0, %1, off"
               :: "v"(l), "v"(gSrc) : "memory");
}
__device__ inline void async_copy_b64(void* ldsDst, const void* gSrc) {
  unsigned l = (unsigned)(uintptr_t)ldsDst;
  asm volatile("global_load_async_to_lds_b64 %0, %1, off"
               :: "v"(l), "v"(gSrc) : "memory");
}
__device__ inline void wait_async0() {
  asm volatile("s_wait_asynccnt 0x0" ::: "memory");
}

// ---------------------------------------------------------------------------
// Weight pre-pack kernels: fp32 -> bf16, padded, in WMMA B-tile LDS layout
// packed[kt][nt][16 n][32 k]
// ---------------------------------------------------------------------------
__global__ __launch_bounds__(256) void pack_leaf_B(const float* __restrict__ Wcx,
                                                   const float* __restrict__ Wox,
                                                   __bf16* __restrict__ dst) {
  int idx = blockIdx.x * 256 + threadIdx.x;       // 10*20*512 = 102400 total
  int k  = idx & 31;
  int n  = (idx >> 5) & 15;
  int nt = (idx >> 9) % NT_LEAF;
  int kt = idx / (NT_LEAF * 512);
  int kg = kt * 32 + k;
  float v = 0.0f;
  if (nt < 10) {
    int col = nt * 16 + n;
    if (kg < EMBED && col < HID) v = Wcx[kg * HID + col];
  } else {
    int col = (nt - 10) * 16 + n;
    if (kg < EMBED && col < HID) v = Wox[kg * HID + col];
  }
  dst[idx] = (__bf16)v;
}

__global__ __launch_bounds__(256) void pack_comp_B(const float* __restrict__ Wl,
                                                   const float* __restrict__ Wr,
                                                   __bf16* __restrict__ dst) {
  int idx = blockIdx.x * 256 + threadIdx.x;       // 10*40*512 = 204800 total
  int k  = idx & 31;
  int n  = (idx >> 5) & 15;
  int nt = (idx >> 9) % NT_COMP;
  int kt = idx / (NT_COMP * 512);
  int kg = kt * 32 + k;                            // 0..319: [0,160)=left h, [160,320)=right h
  int ng = nt * 16 + n;                            // 0..639
  int g  = ng / HPAD;                              // gate
  int kc = ng % HPAD;                              // output column
  float v = 0.0f;
  if (kc < HID) {
    if (kg < HID)                           v = Wl[(g * HID + kg) * HID + kc];
    else if (kg >= HPAD && kg < HPAD + HID) v = Wr[(g * HID + (kg - HPAD)) * HID + kc];
  }
  dst[idx] = (__bf16)v;
}

// ---------------------------------------------------------------------------
// Leaf GEMM: out[M,320] = bf16(X[M,300]) @ B[320,320], fused leaf-LSTM epilogue
// block tile: 64 M x 320 N ; 8 waves = 4(M) x 2(N); 10 accum frags per wave
// ---------------------------------------------------------------------------
__global__ __launch_bounds__(256) void leaf_gemm(const float* __restrict__ X,
                                                 const __bf16* __restrict__ Bp,
                                                 const float* __restrict__ bcx,
                                                 const float* __restrict__ box,
                                                 float* __restrict__ cOut,
                                                 __bf16* __restrict__ hOut) {
  __shared__ __align__(16) unsigned char smem[81920];
  __bf16* sA   = (__bf16*)smem;                 // [64][32]  bf16 (4KB)
  __bf16* sB   = (__bf16*)(smem + 4096);        // [20][16][32] bf16 (20KB)
  float*  sAcc = (float*)smem;                  // [64][320] f32 (80KB, aliased)

  const int tid  = threadIdx.x;
  const int lane = tid & 31;
  const int w    = tid >> 5;
  const int mw   = w & 3;                       // M strip 0..3
  const int nw   = w >> 2;                      // N half 0..1
  const int m0   = blockIdx.x * 64;

  v8f acc[10];
#pragma unroll
  for (int j = 0; j < 10; ++j)
#pragma unroll
    for (int r = 0; r < 8; ++r) acc[j][r] = 0.0f;

  for (int kt = 0; kt < KT_LEAF; ++kt) {
    __syncthreads();
    { // stage B (async, no conversion): 20KB pre-packed bf16 tile
      const char* src = (const char*)(Bp + (size_t)kt * (NT_LEAF * 512));
      char* dst = (char*)sB;
#pragma unroll
      for (int i = 0; i < 5; ++i) {
        unsigned o = (unsigned)(tid + i * 256) * 16u;
        async_copy_b128(dst + o, src + o);
      }
    }
    { // stage A: 64 rows x 32 cols, fp32 -> bf16 (needs VGPR conversion)
      int row = tid >> 2;
      int c0  = (tid & 3) * 8;
      int gk  = kt * 32 + c0;
      const float* src = X + (size_t)(m0 + row) * EMBED + gk;
      __align__(16) __bf16 tmp[8];
#pragma unroll
      for (int i = 0; i < 8; ++i) {
        float v = (gk + i < EMBED) ? src[i] : 0.0f;
        tmp[i] = (__bf16)v;
      }
      *(u32x4*)(sA + row * 32 + c0) = *(const u32x4*)tmp;
    }
    wait_async0();
    __syncthreads();
    FragB16 a = load_fragA(sA + mw * 16 * 32, lane);
#pragma unroll
    for (int j = 0; j < 10; ++j) {
      FragB16 b = load_fragB(sB + (nw * 10 + j) * 512, lane);
      acc[j] = __builtin_amdgcn_wmma_f32_16x16x32_bf16(false, a.bf, false, b.bf,
                                                       (short)0, acc[j], false, false);
    }
  }
  __syncthreads();
  { // spill accumulators: row = base + r + 8*(lane>>4), col = l&15 within n-tile
    int rbase = mw * 16 + (lane >> 4) * 8;
    int col0  = nw * 160 + (lane & 15);
#pragma unroll
    for (int j = 0; j < 10; ++j)
#pragma unroll
      for (int r = 0; r < 8; ++r)
        sAcc[(rbase + r) * 320 + col0 + j * 16] = acc[j][r];
  }
  __syncthreads();
  // epilogue: c = craw + bcx ; h = sigmoid(oraw + box) * tanh(c); zero-pad cols
  for (int it = 0; it < 40; ++it) {
    int idx  = tid + it * 256;                  // 64*160 outputs
    int m    = idx / HPAD;
    int colk = idx % HPAD;
    float cv = 0.0f, hv = 0.0f;
    if (colk < HID) {
      float craw = sAcc[m * 320 + colk] + bcx[colk];
      float oraw = sAcc[m * 320 + 160 + colk] + box[colk];
      cv = craw;
      hv = sigf(oraw) * tanhf(craw);
    }
    size_t gi = (size_t)(m0 + m) * HPAD + colk;
    cOut[gi] = cv;
    hOut[gi] = (__bf16)hv;
  }
}

// ---------------------------------------------------------------------------
// Composer GEMM: A[M,320] = hPrev reinterpreted (rows 2m,2m+1 concatenated),
// out[M,640] = A @ B, fused gate epilogue.
// block tile: 32 M x 640 N ; 8 waves = 2(M) x 4(N); 10 accum frags per wave
// ---------------------------------------------------------------------------
__global__ __launch_bounds__(256) void comp_gemm(const __bf16* __restrict__ hPrev,
                                                 const float* __restrict__ cPrev,
                                                 const __bf16* __restrict__ Bp,
                                                 const float* __restrict__ bl,
                                                 const float* __restrict__ br,
                                                 float* __restrict__ cOut,
                                                 __bf16* __restrict__ hOut,
                                                 int Mout) {
  __shared__ __align__(16) unsigned char smem[81920];
  __bf16* sA   = (__bf16*)smem;                 // [32][32] bf16 (2KB)
  __bf16* sB   = (__bf16*)(smem + 2048);        // [40][16][32] bf16 (40KB)
  float*  sAcc = (float*)smem;                  // [32][640] f32 (80KB, aliased)

  const int tid  = threadIdx.x;
  const int lane = tid & 31;
  const int w    = tid >> 5;
  const int mw   = w & 1;                       // M strip 0..1
  const int nw   = w >> 1;                      // N quarter 0..3 (one gate each)
  const int m0   = blockIdx.x * 32;

  v8f acc[10];
#pragma unroll
  for (int j = 0; j < 10; ++j)
#pragma unroll
    for (int r = 0; r < 8; ++r) acc[j][r] = 0.0f;

  for (int kt = 0; kt < KT_COMP; ++kt) {
    __syncthreads();
    { // stage A (async): 32 rows x 32 bf16 (hPrev viewed as [Mout][320])
      int row = tid >> 3;
      int c4  = (tid & 7) * 4;
      const __bf16* src = hPrev + (size_t)(m0 + row) * 320 + kt * 32 + c4;
      async_copy_b64(sA + row * 32 + c4, src);
    }
    { // stage B (async): 40KB pre-packed bf16 tile
      const char* src = (const char*)(Bp + (size_t)kt * (NT_COMP * 512));
      char* dst = (char*)sB;
#pragma unroll
      for (int i = 0; i < 10; ++i) {
        unsigned o = (unsigned)(tid + i * 256) * 16u;
        async_copy_b128(dst + o, src + o);
      }
    }
    wait_async0();
    __syncthreads();
    FragB16 a = load_fragA(sA + mw * 16 * 32, lane);
#pragma unroll
    for (int j = 0; j < 10; ++j) {
      FragB16 b = load_fragB(sB + (nw * 10 + j) * 512, lane);
      acc[j] = __builtin_amdgcn_wmma_f32_16x16x32_bf16(false, a.bf, false, b.bf,
                                                       (short)0, acc[j], false, false);
    }
  }
  __syncthreads();
  {
    int rbase = mw * 16 + (lane >> 4) * 8;
    int col0  = nw * 160 + (lane & 15);
#pragma unroll
    for (int j = 0; j < 10; ++j)
#pragma unroll
      for (int r = 0; r < 8; ++r)
        sAcc[(rbase + r) * 640 + col0 + j * 16] = acc[j][r];
  }
  __syncthreads();
  // epilogue: gates (i, lf, rf, u), c = i*u + lf*lc + rf*rc, h = tanh(c)
  for (int it = 0; it < 20; ++it) {
    int idx  = tid + it * 256;                  // 32*160 outputs
    int m    = idx / HPAD;
    int colk = idx % HPAD;
    int mg   = m0 + m;
    if (mg >= Mout) continue;
    float cv = 0.0f, hv = 0.0f;
    if (colk < HID) {
      float g0 = sAcc[m * 640 +   0 + colk] + bl[        colk] + br[        colk];
      float g1 = sAcc[m * 640 + 160 + colk] + bl[HID   + colk] + br[HID   + colk];
      float g2 = sAcc[m * 640 + 320 + colk] + bl[2*HID + colk] + br[2*HID + colk];
      float g3 = sAcc[m * 640 + 480 + colk] + bl[3*HID + colk] + br[3*HID + colk];
      float ig = sigf(g0), lf = sigf(g1), rf = sigf(g2), u = tanhf(g3);
      float lc = cPrev[(size_t)(2 * mg)     * HPAD + colk];
      float rc = cPrev[(size_t)(2 * mg + 1) * HPAD + colk];
      cv = ig * u + lf * lc + rf * rc;
      hv = tanhf(cv);
    }
    size_t gi = (size_t)mg * HPAD + colk;
    cOut[gi] = cv;
    hOut[gi] = (__bf16)hv;
  }
}

// ---------------------------------------------------------------------------
// Classifier + per-node NLL (NLAB=5 -> plain VALU). Root launch writes logits.
// ---------------------------------------------------------------------------
__global__ __launch_bounds__(256) void classifier(const __bf16* __restrict__ h,
                                                  const int* __restrict__ labels,
                                                  const float* __restrict__ Wout,
                                                  const float* __restrict__ bout,
                                                  float* __restrict__ nll,
                                                  float* __restrict__ outLogits,
                                                  int M, int nodeOff) {
  int n = blockIdx.x * 256 + threadIdx.x;
  if (n >= M) return;
  const __bf16* hr = h + (size_t)n * HPAD;
  float lg[NLAB];
#pragma unroll
  for (int c = 0; c < NLAB; ++c) lg[c] = bout[c];
  for (int k = 0; k < HID; ++k) {
    float hv = (float)hr[k];
#pragma unroll
    for (int c = 0; c < NLAB; ++c) lg[c] += hv * Wout[k * NLAB + c];
  }
  float mx = lg[0];
#pragma unroll
  for (int c = 1; c < NLAB; ++c) mx = fmaxf(mx, lg[c]);
  float s = 0.0f;
#pragma unroll
  for (int c = 0; c < NLAB; ++c) s += __expf(lg[c] - mx);
  float lse = __logf(s) + mx;
  int lab = labels[nodeOff + n];
  nll[nodeOff + n] = lse - lg[lab];
  if (M == 1 && n == 0) {
#pragma unroll
    for (int c = 0; c < NLAB; ++c) outLogits[c] = lg[c];
  }
}

// Deterministic fixed-order reduction (single block)
__global__ __launch_bounds__(256) void reduce_loss(const float* __restrict__ nll,
                                                   int n, float* __restrict__ out) {
  __shared__ float sbuf[256];
  float s = 0.0f;
  for (int i = threadIdx.x; i < n; i += 256) s += nll[i];
  sbuf[threadIdx.x] = s;
  __syncthreads();
  for (int st = 128; st > 0; st >>= 1) {
    if (threadIdx.x < st) sbuf[threadIdx.x] += sbuf[threadIdx.x + st];
    __syncthreads();
  }
  if (threadIdx.x == 0) out[NLAB] = sbuf[0];
}

// ---------------------------------------------------------------------------
// Launch
// ---------------------------------------------------------------------------
extern "C" void kernel_launch(void* const* d_in, const int* in_sizes, int n_in,
                              void* d_out, int out_size, void* d_ws, size_t ws_size,
                              hipStream_t stream) {
  const float* embeds = (const float*)d_in[0];
  const int*   labels = (const int*)d_in[1];
  const float* Wcx = (const float*)d_in[2];
  const float* bcx = (const float*)d_in[3];
  const float* Wox = (const float*)d_in[4];
  const float* box = (const float*)d_in[5];
  const float* Wl  = (const float*)d_in[6];
  const float* bl  = (const float*)d_in[7];
  const float* Wr  = (const float*)d_in[8];
  const float* br  = (const float*)d_in[9];
  const float* Wout = (const float*)d_in[10];
  const float* bout = (const float*)d_in[11];
  float* out = (float*)d_out;

  char* ws = (char*)d_ws;
  size_t off = 0;
  auto alloc = [&](size_t bytes) {
    char* p = ws + off;
    off = (off + bytes + 255) & ~(size_t)255;
    return p;
  };
  __bf16* packLeaf = (__bf16*)alloc((size_t)KT_LEAF * NT_LEAF * 512 * 2);
  __bf16* packComp = (__bf16*)alloc((size_t)KT_COMP * NT_COMP * 512 * 2);
  float*  cA = (float*)alloc((size_t)N_LEAVES * HPAD * 4);
  float*  cB = (float*)alloc((size_t)(N_LEAVES / 2) * HPAD * 4);
  __bf16* hA = (__bf16*)alloc((size_t)N_LEAVES * HPAD * 2);
  __bf16* hB = (__bf16*)alloc((size_t)(N_LEAVES / 2) * HPAD * 2);
  float*  nll = (float*)alloc((size_t)(2 * N_LEAVES - 1) * 4 + 65536);

  pack_leaf_B<<<(KT_LEAF * NT_LEAF * 512) / 256, 256, 0, stream>>>(Wcx, Wox, packLeaf);
  pack_comp_B<<<(KT_COMP * NT_COMP * 512) / 256, 256, 0, stream>>>(Wl, Wr, packComp);

  // leaf level
  leaf_gemm<<<N_LEAVES / 64, 256, 0, stream>>>(embeds, packLeaf, bcx, box, cA, hA);
  classifier<<<N_LEAVES / 256, 256, 0, stream>>>(hA, labels, Wout, bout, nll, out,
                                                 N_LEAVES, 0);

  // level-wise composition up the tree (ping-pong c/h buffers)
  int nodeOff = N_LEAVES;
  float*  cp = cA; __bf16* hp = hA;
  float*  cn = cB; __bf16* hn = hB;
  for (int M = N_LEAVES / 2; M >= 1; M >>= 1) {
    comp_gemm<<<(M + 31) / 32, 256, 0, stream>>>(hp, cp, packComp, bl, br, cn, hn, M);
    classifier<<<(M + 255) / 256, 256, 0, stream>>>(hn, labels, Wout, bout, nll, out,
                                                    M, nodeOff);
    nodeOff += M;
    float* tc = cp; cp = cn; cn = tc;
    __bf16* th = hp; hp = hn; hn = th;
  }

  reduce_loss<<<1, 256, 0, stream>>>(nll, 2 * N_LEAVES - 1, out);
}
